// MultiHeadAttention_40767829574475
// MI455X (gfx1250) — matrix operand order repaired and checked by reading, not compiled, not run on hardware
//
#include <hip/hip_runtime.h>
#include <stdint.h>

#define DEV __device__ __forceinline__

typedef __attribute__((ext_vector_type(16))) __bf16 bf16x16;
typedef __attribute__((ext_vector_type(4)))  __bf16 bf16x4;
typedef __attribute__((ext_vector_type(8)))  float  f32x8;

enum { kB = 4, kS = 2048, kD = 1024, kH = 16, kHD = 64 };
enum { kLdT = 40 };  // padded K-stride for transposed LDS tiles (multiple of 8)

// ---------------------------------------------------------------------------
// CDNA5 async global->LDS staging (ASYNCcnt-tracked), GVS addressing mode.
// ---------------------------------------------------------------------------
DEV void async_load_b128(uint32_t lds_byte_off, uint64_t gbase,
                         uint32_t voff_bytes) {
  asm volatile("global_load_async_to_lds_b128 %0, %1, %2"
               :: "v"(lds_byte_off), "v"(voff_bytes), "s"(gbase)
               : "memory");
}
template <int N>
DEV void wait_async() {
  asm volatile("s_wait_asynccnt %0" :: "i"(N) : "memory");
}
DEV uint32_t lds_off(const void* p) { return (uint32_t)(uintptr_t)p; }

// ---------------------------------------------------------------------------
// WMMA fragment helpers (this code's contiguous k = 16*(lane/16) + e order, wave32)
// ---------------------------------------------------------------------------
DEV bf16x16 load_a_frag(const __bf16* p, int ldm, int lane) {
  const int hl = lane >> 4, row = lane & 15;
  const __bf16* r = p + row * ldm + hl * 16;
  bf16x16 a;
#pragma unroll
  for (int e = 0; e < 16; ++e) a[e] = r[e];
  return a;
}

// logical B[k][n] = src[n*ld + k]; 16 contiguous bf16 per lane (2x b128).
DEV bf16x16 load_bt_frag(const __bf16* p, int ld, int lane) {
  const int hl = lane >> 4, col = lane & 15;
  const __bf16* r = p + (size_t)col * ld + hl * 16;
  bf16x16 b;
#pragma unroll
  for (int e = 0; e < 16; ++e) b[e] = r[e];
  return b;
}

DEV f32x8 wmma_bf16(bf16x16 a, bf16x16 b, f32x8 c) {
  return __builtin_amdgcn_wmma_f32_16x16x32_bf16(false, a, false, b, (short)0,
                                                 c, false, false);
}

DEV void zero8(f32x8& v) {
#pragma unroll
  for (int j = 0; j < 8; ++j) v[j] = 0.0f;
}

// Split register-mediated transpose staging for 32x64 row-major tiles.
DEV uint4 ldg_tile_row(const __bf16* __restrict__ gsrc, size_t ld, int tid) {
  const int row = tid >> 3, q = tid & 7;
  return *reinterpret_cast<const uint4*>(gsrc + (size_t)row * ld + q * 8);
}
DEV void scatter_T(uint4 u, __bf16* dstT, int tid) {
  const int row = tid >> 3, q = tid & 7;
  union { uint4 u; __bf16 h[8]; } wv;
  wv.u = u;
#pragma unroll
  for (int j = 0; j < 8; ++j) dstT[(q * 8 + j) * kLdT + row] = wv.h[j];
}

// ---------------------------------------------------------------------------
// Kernel: f32 -> bf16 conversion (vectorized)
// ---------------------------------------------------------------------------
__global__ void f32_to_bf16_kernel(const float* __restrict__ in,
                                   __bf16* __restrict__ out, int n4) {
  int i = blockIdx.x * blockDim.x + threadIdx.x;
  if (i >= n4) return;
  float4 v = reinterpret_cast<const float4*>(in)[i];
  bf16x4 o;
  o[0] = (__bf16)v.x; o[1] = (__bf16)v.y;
  o[2] = (__bf16)v.z; o[3] = (__bf16)v.w;
  reinterpret_cast<bf16x4*>(out)[i] = o;
}

// ---------------------------------------------------------------------------
// Kernel: per-head QKV projection.  grid = (S/256, B*H), block = 256 (8 waves)
// Each wave: 32 rows x 64 cols (2 A-frags reuse 4 B-frags -> 8 WMMA / K-step).
// Double-buffered async A staging + register-prefetched W transpose staging.
// ---------------------------------------------------------------------------
__global__ __launch_bounds__(256) void qkv_proj_kernel(
    const __bf16* __restrict__ xb,   // [B*S, D]
    const __bf16* __restrict__ Wb,   // [H, D, HD]
    const float* __restrict__ bias,  // [H, HD]
    float scale,                     // folded softmax scale (Q only)
    __bf16* __restrict__ outp) {     // [B, H, S, HD]
  __shared__ alignas(16) __bf16 xsb[2][256 * 32];    // A tiles (ping-pong)
  __shared__ alignas(16) __bf16 wTb[2][64 * kLdT];   // B tiles transposed
  const int tid = threadIdx.x, wave = tid >> 5, lane = tid & 31;
  const int tile = blockIdx.x, bh = blockIdx.y;
  const int b = bh / kH, h = bh % kH;

  f32x8 acc[2][4];
#pragma unroll
  for (int g = 0; g < 2; ++g)
#pragma unroll
    for (int n = 0; n < 4; ++n) zero8(acc[g][n]);

  const __bf16* xrow = xb + (size_t)(b * kS + tile * 256) * kD;
  const __bf16* Wh = Wb + (size_t)h * kD * kHD;

  auto issue_xs = [&](int buf, int kk) {
    const uint32_t base = lds_off(xsb[buf]);
    const uint64_t gx = (uint64_t)(uintptr_t)(xrow + kk);
#pragma unroll
    for (int j = 0; j < 4; ++j) {
      int idx = tid + 256 * j;
      int row = idx >> 2, quad = idx & 3;
      async_load_b128(base + idx * 16, gx,
                      (uint32_t)((row * kD + quad * 8) * 2));
    }
  };

  issue_xs(0, 0);
  uint4 wreg = ldg_tile_row(Wh, kHD, tid);

  const int kSteps = kD / 32;
  for (int s = 0; s < kSteps; ++s) {
    const int cur = s & 1;
    uint4 wnext;
    if (s + 1 < kSteps) {
      issue_xs(cur ^ 1, (s + 1) * 32);
      wnext = ldg_tile_row(Wh + (size_t)(s + 1) * 32 * kHD, kHD, tid);
    }
    scatter_T(wreg, wTb[cur], tid);
    if (s + 1 < kSteps) wait_async<4>(); else wait_async<0>();
    __syncthreads();

    bf16x16 a0 = load_a_frag(xsb[cur] + (wave * 32) * 32, 32, lane);
    bf16x16 a1 = load_a_frag(xsb[cur] + (wave * 32 + 16) * 32, 32, lane);
#pragma unroll
    for (int n = 0; n < 4; ++n) {
      bf16x16 bf = load_bt_frag(wTb[cur] + n * 16 * kLdT, kLdT, lane);
      acc[0][n] = wmma_bf16(a0, bf, acc[0][n]);
      acc[1][n] = wmma_bf16(a1, bf, acc[1][n]);
    }
    __syncthreads();
    wreg = wnext;
  }

  const int hl = lane >> 4, col = lane & 15;
  const float* bvec = bias + h * kHD;
#pragma unroll
  for (int g = 0; g < 2; ++g) {
    const int row0 = tile * 256 + wave * 32 + g * 16;
    __bf16* dst = outp + ((size_t)bh * kS + row0) * kHD;
#pragma unroll
    for (int n = 0; n < 4; ++n)
#pragma unroll
      for (int i = 0; i < 8; ++i) {
        int r = i + hl * 8, c = n * 16 + col;
        dst[(size_t)r * kHD + c] = (__bf16)((acc[g][n][i] + bvec[c]) * scale);
      }
  }
}

// ---------------------------------------------------------------------------
// Kernel: flash attention over one head.  grid = (S/128, B*H), block = 256.
// Q pre-scaled by 1/sqrt(HD).  Ping-ponged async K tiles + transposed V tiles.
// ---------------------------------------------------------------------------
__global__ __launch_bounds__(256) void flash_attn_kernel(
    const __bf16* __restrict__ Qb, const __bf16* __restrict__ Kb,
    const __bf16* __restrict__ Vb, __bf16* __restrict__ Ab) {
  __shared__ alignas(16) __bf16 smem[13312];  // 26 KB
  // layout: kvK ping-pong @0 (2x2048), vT ping-pong @4096 (2x2560),
  //         per-wave P tiles @9216 (8x512)
  const int tid = threadIdx.x, wave = tid >> 5, lane = tid & 31;
  __bf16* pbuf = smem + 9216 + wave * 512;       // per-wave 16x32 P tile
  const int tile = blockIdx.x, bh = blockIdx.y;
  const int b = bh / kH, h = bh % kH;
  const int hl = lane >> 4, col = lane & 15;
  const size_t head_off = (size_t)bh * kS * kHD;

  // async-stage this block's Q tile (128x64 contiguous) and pull wave frags
  {
    const uint64_t gq =
        (uint64_t)(uintptr_t)(Qb + head_off + (size_t)tile * 128 * kHD);
    const uint32_t qs = lds_off(smem);
#pragma unroll
    for (int j = 0; j < 4; ++j) {
      uint32_t o = (uint32_t)(tid + 256 * j) * 16u;
      async_load_b128(qs + o, gq, o);
    }
  }
  wait_async<0>();
  __syncthreads();
  bf16x16 aq0 = load_a_frag(smem + wave * 16 * 64, 64, lane);
  bf16x16 aq1 = load_a_frag(smem + wave * 16 * 64 + 32, 64, lane);
  __syncthreads();

  float mrow[8], lrow[8];
#pragma unroll
  for (int i = 0; i < 8; ++i) { mrow[i] = -3.0e38f; lrow[i] = 0.0f; }
  f32x8 o[4];
#pragma unroll
  for (int n = 0; n < 4; ++n) zero8(o[n]);

  const __bf16* ksrc = Kb + head_off;
  const __bf16* vsrc = Vb + head_off;

  // prologue: chunk 0 in flight
  async_load_b128(lds_off(smem) + (uint32_t)tid * 16u,
                  (uint64_t)(uintptr_t)ksrc, (uint32_t)tid * 16u);
  uint4 vreg = ldg_tile_row(vsrc, kHD, tid);

  const int nChunks = kS / 32;
  for (int c = 0; c < nChunks; ++c) {
    const int cur = c & 1;
    __bf16* kCur = smem + cur * 2048;
    __bf16* vCur = smem + 4096 + cur * 2560;
    uint4 vnext;
    if (c + 1 < nChunks) {
      async_load_b128(lds_off(smem) + (uint32_t)((cur ^ 1) * 4096) +
                          (uint32_t)tid * 16u,
                      (uint64_t)(uintptr_t)(ksrc + (size_t)(c + 1) * 32 * kHD),
                      (uint32_t)tid * 16u);
      vnext = ldg_tile_row(vsrc + (size_t)(c + 1) * 32 * kHD, kHD, tid);
    }
    scatter_T(vreg, vCur, tid);
    if (c + 1 < nChunks) wait_async<1>(); else wait_async<0>();
    __syncthreads();

    // scores: 16 q-rows x 32 kv-rows (Q pre-scaled by 1/8)
    f32x8 s0, s1;
    zero8(s0); zero8(s1);
    s0 = wmma_bf16(aq0, load_bt_frag(kCur, 64, lane), s0);
    s0 = wmma_bf16(aq1, load_bt_frag(kCur + 32, 64, lane), s0);
    s1 = wmma_bf16(aq0, load_bt_frag(kCur + 16 * 64, 64, lane), s1);
    s1 = wmma_bf16(aq1, load_bt_frag(kCur + 16 * 64 + 32, 64, lane), s1);

    // online softmax; a row's 16 lane-values live within one 16-lane half
#pragma unroll
    for (int i = 0; i < 8; ++i) {
      float v0 = s0[i], v1 = s1[i];
      float cm = fmaxf(v0, v1);
#pragma unroll
      for (int d = 1; d < 16; d <<= 1) cm = fmaxf(cm, __shfl_xor(cm, d, 32));
      float mnew = fmaxf(mrow[i], cm);
      float p0 = __expf(v0 - mnew);
      float p1 = __expf(v1 - mnew);
      float corr = __expf(mrow[i] - mnew);
      mrow[i] = mnew;
      float ps = p0 + p1;
#pragma unroll
      for (int d = 1; d < 16; d <<= 1) ps += __shfl_xor(ps, d, 32);
      lrow[i] = lrow[i] * corr + ps;
      pbuf[(i + hl * 8) * 32 + col] = (__bf16)p0;
      pbuf[(i + hl * 8) * 32 + 16 + col] = (__bf16)p1;
#pragma unroll
      for (int n = 0; n < 4; ++n) o[n][i] *= corr;
    }

    // O += P (16x32, via per-wave LDS relayout) x V (32x64)
    bf16x16 ap = load_a_frag(pbuf, 32, lane);
#pragma unroll
    for (int n = 0; n < 4; ++n)
      o[n] = wmma_bf16(ap, load_bt_frag(vCur + n * 16 * kLdT, kLdT, lane),
                       o[n]);
    __syncthreads();
    vreg = vnext;
  }

  const int row0 = tile * 128 + wave * 16;
  __bf16* dst = Ab + (size_t)(b * kS + row0) * kD + h * kHD;
#pragma unroll
  for (int i = 0; i < 8; ++i) {
    float inv = 1.0f / lrow[i];
#pragma unroll
    for (int n = 0; n < 4; ++n)
      dst[(size_t)(i + hl * 8) * kD + n * 16 + col] = (__bf16)(o[n][i] * inv);
  }
}

// ---------------------------------------------------------------------------
// Kernel: output projection [B*S,D] x [D,D] + bo -> f32.
// grid = (B*S/256, D/64), block = 256 (8 waves), 32-row per-wave tiles.
// ---------------------------------------------------------------------------
__global__ __launch_bounds__(256) void out_proj_kernel(
    const __bf16* __restrict__ Ain,  // [B*S, D]
    const __bf16* __restrict__ Wob,  // [D, D]
    const float* __restrict__ bo,    // [D]
    float* __restrict__ outp) {      // [B*S, D]
  __shared__ alignas(16) __bf16 xsb[2][256 * 32];
  __shared__ alignas(16) __bf16 wTb[2][64 * kLdT];
  const int tid = threadIdx.x, wave = tid >> 5, lane = tid & 31;
  const int rtile = blockIdx.x, n0 = blockIdx.y * 64;

  f32x8 acc[2][4];
#pragma unroll
  for (int g = 0; g < 2; ++g)
#pragma unroll
    for (int n = 0; n < 4; ++n) zero8(acc[g][n]);

  const __bf16* arow = Ain + (size_t)rtile * 256 * kD;

  auto issue_xs = [&](int buf, int kk) {
    const uint32_t base = lds_off(xsb[buf]);
    const uint64_t ga = (uint64_t)(uintptr_t)(arow + kk);
#pragma unroll
    for (int j = 0; j < 4; ++j) {
      int idx = tid + 256 * j;
      int row = idx >> 2, quad = idx & 3;
      async_load_b128(base + idx * 16, ga,
                      (uint32_t)((row * kD + quad * 8) * 2));
    }
  };

  issue_xs(0, 0);
  uint4 wreg = ldg_tile_row(Wob + n0, kD, tid);

  const int kSteps = kD / 32;
  for (int s = 0; s < kSteps; ++s) {
    const int cur = s & 1;
    uint4 wnext;
    if (s + 1 < kSteps) {
      issue_xs(cur ^ 1, (s + 1) * 32);
      wnext = ldg_tile_row(Wob + (size_t)(s + 1) * 32 * kD + n0, kD, tid);
    }
    scatter_T(wreg, wTb[cur], tid);
    if (s + 1 < kSteps) wait_async<4>(); else wait_async<0>();
    __syncthreads();

    bf16x16 a0 = load_a_frag(xsb[cur] + (wave * 32) * 32, 32, lane);
    bf16x16 a1 = load_a_frag(xsb[cur] + (wave * 32 + 16) * 32, 32, lane);
#pragma unroll
    for (int n = 0; n < 4; ++n) {
      bf16x16 bf = load_bt_frag(wTb[cur] + n * 16 * kLdT, kLdT, lane);
      acc[0][n] = wmma_bf16(a0, bf, acc[0][n]);
      acc[1][n] = wmma_bf16(a1, bf, acc[1][n]);
    }
    __syncthreads();
    wreg = wnext;
  }

  const int hl = lane >> 4, col = lane & 15;
#pragma unroll
  for (int g = 0; g < 2; ++g) {
    const int row0 = rtile * 256 + wave * 32 + g * 16;
#pragma unroll
    for (int n = 0; n < 4; ++n)
#pragma unroll
      for (int i = 0; i < 8; ++i) {
        int c = n0 + n * 16 + col;
        outp[(size_t)(row0 + i + hl * 8) * kD + c] = acc[g][n][i] + bo[c];
      }
  }
}

// ---------------------------------------------------------------------------
// Host launch
// ---------------------------------------------------------------------------
extern "C" void kernel_launch(void* const* d_in, const int* in_sizes, int n_in,
                              void* d_out, int out_size, void* d_ws,
                              size_t ws_size, hipStream_t stream) {
  (void)in_sizes; (void)n_in; (void)out_size; (void)ws_size;

  const float* x  = (const float*)d_in[0];
  const float* Wq = (const float*)d_in[1];
  const float* bq = (const float*)d_in[2];
  const float* Wk = (const float*)d_in[3];
  const float* bk = (const float*)d_in[4];
  const float* Wv = (const float*)d_in[5];
  const float* bv = (const float*)d_in[6];
  const float* Wo = (const float*)d_in[7];
  const float* bo = (const float*)d_in[8];
  float* out = (float*)d_out;

  char* wsp = (char*)d_ws;
  size_t off = 0;
  auto alloc = [&](size_t bytes) {
    char* p = wsp + off;
    off += (bytes + 255) & ~(size_t)255;
    return p;
  };
  __bf16* xb   = (__bf16*)alloc((size_t)kB * kS * kD * 2);
  __bf16* Wqb  = (__bf16*)alloc((size_t)kH * kD * kHD * 2);
  __bf16* Wkb  = (__bf16*)alloc((size_t)kH * kD * kHD * 2);
  __bf16* Wvb  = (__bf16*)alloc((size_t)kH * kD * kHD * 2);
  __bf16* Wob  = (__bf16*)alloc((size_t)kD * kD * 2);
  __bf16* Qb   = (__bf16*)alloc((size_t)kB * kH * kS * kHD * 2);
  __bf16* Kb   = (__bf16*)alloc((size_t)kB * kH * kS * kHD * 2);
  __bf16* Vb   = (__bf16*)alloc((size_t)kB * kH * kS * kHD * 2);
  __bf16* Abuf = (__bf16*)alloc((size_t)kB * kS * kD * 2);

  auto conv = [&](const float* src, __bf16* dst, size_t n) {
    int n4 = (int)(n / 4);
    f32_to_bf16_kernel<<<(n4 + 255) / 256, 256, 0, stream>>>(src, dst, n4);
  };
  conv(x,  xb,  (size_t)kB * kS * kD);
  conv(Wq, Wqb, (size_t)kH * kD * kHD);
  conv(Wk, Wkb, (size_t)kH * kD * kHD);
  conv(Wv, Wvb, (size_t)kH * kD * kHD);
  conv(Wo, Wob, (size_t)kD * kD);

  const float qscale = 0.125f;  // 1/sqrt(HD), folded into Q projection
  dim3 gProj(kS / 256, kB * kH);
  qkv_proj_kernel<<<gProj, 256, 0, stream>>>(xb, Wqb, bq, qscale, Qb);
  qkv_proj_kernel<<<gProj, 256, 0, stream>>>(xb, Wkb, bk, 1.0f, Kb);
  qkv_proj_kernel<<<gProj, 256, 0, stream>>>(xb, Wvb, bv, 1.0f, Vb);

  dim3 gAttn(kS / 128, kB * kH);
  flash_attn_kernel<<<gAttn, 256, 0, stream>>>(Qb, Kb, Vb, Abuf);

  dim3 gOut((kB * kS) / 256, kD / 64);
  out_proj_kernel<<<gOut, 256, 0, stream>>>(Abuf, Wob, bo, out);
}
